// Gemma3Attention_85890755985607
// MI455X (gfx1250) — matrix operand order repaired and checked by reading, not compiled, not run on hardware
//
#include <hip/hip_runtime.h>
#include <hip/hip_bf16.h>

#define S_LEN 2048
#define H_DIM 2560
#define NH 8
#define NKV 4
#define HD 256
#define QKV_N ((NH + 2 * NKV) * HD)   // 4096
#define ATTN_N (NH * HD)              // 2048
#define SCALING 0.0625f               // 256^-0.5
#define WINDOW 1023
#define NEG_BIG (-1.0e30f)

typedef __attribute__((ext_vector_type(16))) __bf16 v16bf;
typedef __attribute__((ext_vector_type(8)))  float  v8f;
typedef __attribute__((ext_vector_type(4)))  int    i4;

// CDNA5 async global->LDS path (ASYNCcnt), guarded so either toolchain compiles.
#if defined(__AMDGCN__) && __has_builtin(__builtin_amdgcn_global_load_async_to_lds_b128) && __has_builtin(__builtin_amdgcn_s_wait_asynccnt)
#define HAS_ASYNC 1
#else
#define HAS_ASYNC 0
#endif

#if HAS_ASYNC
typedef __attribute__((address_space(1))) i4* g_i4p;   // global int4*
typedef __attribute__((address_space(3))) i4* l_i4p;   // LDS int4*
static __device__ __forceinline__ void async_b128(const void* g, void* l) {
    __builtin_amdgcn_global_load_async_to_lds_b128((g_i4p)g, (l_i4p)l, 0, 0);
}
static __device__ __forceinline__ void wait_async0() {
    __builtin_amdgcn_s_wait_asynccnt(0);
}
#endif

// ---------- helpers ----------
static __device__ __forceinline__ unsigned short f2bf(float f) {
    unsigned int u = __builtin_bit_cast(unsigned int, f);
    u += 0x7fffu + ((u >> 16) & 1u);           // round-to-nearest-even
    return (unsigned short)(u >> 16);
}
static __device__ __forceinline__ __bf16 bfbits(unsigned short b) {
    return __builtin_bit_cast(__bf16, b);
}
static __device__ __forceinline__ void unpack8(uint4 q, v16bf& d, int o) {
    d[o + 0] = bfbits((unsigned short)(q.x & 0xffffu));
    d[o + 1] = bfbits((unsigned short)(q.x >> 16));
    d[o + 2] = bfbits((unsigned short)(q.y & 0xffffu));
    d[o + 3] = bfbits((unsigned short)(q.y >> 16));
    d[o + 4] = bfbits((unsigned short)(q.z & 0xffffu));
    d[o + 5] = bfbits((unsigned short)(q.z >> 16));
    d[o + 6] = bfbits((unsigned short)(q.w & 0xffffu));
    d[o + 7] = bfbits((unsigned short)(q.w >> 16));
}

// ---------- fp32 -> bf16 convert ----------
__global__ void cvt_f32_bf16(const float* __restrict__ in,
                             unsigned short* __restrict__ out, long n4) {
    long i = (long)blockIdx.x * 256 + threadIdx.x;
    if (i < n4) {
        float4 f = ((const float4*)in)[i];
        union { uint2 v; unsigned short u[4]; } t;
        t.u[0] = f2bf(f.x); t.u[1] = f2bf(f.y);
        t.u[2] = f2bf(f.z); t.u[3] = f2bf(f.w);
        ((uint2*)out)[i] = t.v;
    }
}

// ---------- bf16 GEMM: C[M,N] = A[M,K] * B[K,N], fp32 accumulate ----------
// 128x128 block tile, 8 waves = 2(M) x 4(N), wave 64x32 = 4x2 WMMA tiles,
// K-step 32, double-buffered LDS with async global->LDS for A.
__global__ __launch_bounds__(256)
void gemm_bf16(const unsigned short* __restrict__ A,
               const unsigned short* __restrict__ B,
               float* __restrict__ C, int M, int N, int K) {
    __shared__ alignas(16) unsigned short lA[2][128 * 32];   // [m][k]
    __shared__ alignas(16) unsigned short lB[2][128 * 32];   // transposed: [n][k]
    const int tid  = threadIdx.x;
    const int lane = tid & 31;
    const int wv   = tid >> 5;
    const int wm   = wv >> 2;
    const int wn   = wv & 3;
    const int hi   = (lane >> 4) & 1;
    const int ln   = lane & 15;
    const int m0 = blockIdx.y * 128;
    const int n0 = blockIdx.x * 128;
    const int kp = tid >> 4;     // B staging: k row pair 0..15
    const int nc = tid & 15;     // B staging: n chunk 0..15

    v8f c[4][2];
    const v8f vz = {0.f, 0.f, 0.f, 0.f, 0.f, 0.f, 0.f, 0.f};
#pragma unroll
    for (int i = 0; i < 4; ++i)
#pragma unroll
        for (int j = 0; j < 2; ++j) c[i][j] = vz;

    // ---- prologue: stage tile 0 into buffer 0 ----
    {
#if HAS_ASYNC
#pragma unroll
        for (int ch = 0; ch < 2; ++ch) {
            int cid = tid * 2 + ch;
            int row = cid >> 2, cc = cid & 3;
            async_b128(A + (size_t)(m0 + row) * K + cc * 8,
                       &lA[0][row * 32 + cc * 8]);
        }
#else
#pragma unroll
        for (int ch = 0; ch < 2; ++ch) {
            int cid = tid * 2 + ch;
            int row = cid >> 2, cc = cid & 3;
            *(uint4*)&lA[0][row * 32 + cc * 8] =
                *(const uint4*)(A + (size_t)(m0 + row) * K + cc * 8);
        }
#endif
        union { uint4 v; unsigned short u[8]; } t0, t1;
        t0.v = *(const uint4*)(B + (size_t)(2 * kp) * N + n0 + nc * 8);
        t1.v = *(const uint4*)(B + (size_t)(2 * kp + 1) * N + n0 + nc * 8);
#pragma unroll
        for (int j = 0; j < 8; ++j) {
            unsigned int w = (unsigned int)t0.u[j] | ((unsigned int)t1.u[j] << 16);
            *(unsigned int*)&lB[0][(nc * 8 + j) * 32 + 2 * kp] = w;
        }
#if HAS_ASYNC
        wait_async0();
#endif
        __syncthreads();
    }

    const int nk = K / 32;
    for (int ki = 0; ki < nk; ++ki) {
        const int cur = ki & 1, nxt = cur ^ 1;
        const bool have_next = (ki + 1 < nk);
        uint4 rb0, rb1;
#if !HAS_ASYNC
        uint4 ra[2];
#endif
        if (have_next) {
            const int k0n = (ki + 1) * 32;
#if HAS_ASYNC
#pragma unroll
            for (int ch = 0; ch < 2; ++ch) {
                int cid = tid * 2 + ch;
                int row = cid >> 2, cc = cid & 3;
                async_b128(A + (size_t)(m0 + row) * K + k0n + cc * 8,
                           &lA[nxt][row * 32 + cc * 8]);
            }
#else
#pragma unroll
            for (int ch = 0; ch < 2; ++ch) {
                int cid = tid * 2 + ch;
                int row = cid >> 2, cc = cid & 3;
                ra[ch] = *(const uint4*)(A + (size_t)(m0 + row) * K + k0n + cc * 8);
            }
#endif
            rb0 = *(const uint4*)(B + (size_t)(k0n + 2 * kp) * N + n0 + nc * 8);
            rb1 = *(const uint4*)(B + (size_t)(k0n + 2 * kp + 1) * N + n0 + nc * 8);
            if (k0n + 32 < K) {   // prefetch tile after next into near caches
                __builtin_prefetch(A + (size_t)(m0 + (tid >> 1)) * K + k0n + 32, 0, 3);
                __builtin_prefetch(B + (size_t)(k0n + 32 + (tid & 31)) * N + n0 + ((tid >> 5) & 7) * 16, 0, 3);
            }
        }

        // ---- compute on buffer `cur` ----
        v16bf a[4];
#pragma unroll
        for (int mt = 0; mt < 4; ++mt) {
            const unsigned short* p = &lA[cur][(wm * 64 + mt * 16 + ln) * 32 + hi * 16];
            unpack8(*(const uint4*)p, a[mt], 0);
            unpack8(*(const uint4*)(p + 8), a[mt], 8);
        }
#pragma unroll
        for (int nt = 0; nt < 2; ++nt) {
            const unsigned short* p = &lB[cur][(wn * 32 + nt * 16 + ln) * 32 + hi * 16];
            v16bf b;
            unpack8(*(const uint4*)p, b, 0);
            unpack8(*(const uint4*)(p + 8), b, 8);
#pragma unroll
            for (int mt = 0; mt < 4; ++mt)
                c[mt][nt] = __builtin_amdgcn_wmma_f32_16x16x32_bf16(
                    false, a[mt], false, b, (short)0, c[mt][nt], false, false);
        }

        if (have_next) {
#if !HAS_ASYNC
#pragma unroll
            for (int ch = 0; ch < 2; ++ch) {
                int cid = tid * 2 + ch;
                int row = cid >> 2, cc = cid & 3;
                *(uint4*)&lA[nxt][row * 32 + cc * 8] = ra[ch];
            }
#endif
            union { uint4 v; unsigned short u[8]; } t0, t1;
            t0.v = rb0; t1.v = rb1;
#pragma unroll
            for (int j = 0; j < 8; ++j) {
                unsigned int w = (unsigned int)t0.u[j] | ((unsigned int)t1.u[j] << 16);
                *(unsigned int*)&lB[nxt][(nc * 8 + j) * 32 + 2 * kp] = w;
            }
        }
#if HAS_ASYNC
        wait_async0();
#endif
        __syncthreads();
    }

#pragma unroll
    for (int mt = 0; mt < 4; ++mt)
#pragma unroll
        for (int nt = 0; nt < 2; ++nt) {
            int col = n0 + wn * 32 + nt * 16 + ln;
#pragma unroll
            for (int r = 0; r < 8; ++r) {
                int rowm = m0 + wm * 64 + mt * 16 + r + hi * 8;
                C[(size_t)rowm * N + col] = c[mt][nt][r];
            }
        }
}

// ---------- fused RMS-norm + RoPE + bf16 pack ----------
__global__ __launch_bounds__(256)
void normrope_kernel(const float* __restrict__ qkv,
                     const float* __restrict__ cosb, const float* __restrict__ sinb,
                     const float* __restrict__ qw, const float* __restrict__ kw,
                     unsigned short* __restrict__ qo,
                     unsigned short* __restrict__ ko,
                     unsigned short* __restrict__ vo) {
    const int row = blockIdx.x;
    const int s = row >> 4;
    const int h = row & 15;          // block-uniform
    const int d = threadIdx.x;
    float x = qkv[(size_t)s * QKV_N + h * HD + d];

    if (h >= 12) {                   // V: straight convert
        vo[((size_t)(h - 12) * S_LEN + s) * HD + d] = f2bf(x);
        return;
    }
    __shared__ float sh[HD];
    __shared__ float red[8];
    float v = x * x;
#pragma unroll
    for (int o = 16; o > 0; o >>= 1) v += __shfl_xor(v, o, 32);
    if ((threadIdx.x & 31) == 0) red[threadIdx.x >> 5] = v;
    __syncthreads();
    float tot = 0.f;
#pragma unroll
    for (int i = 0; i < 8; ++i) tot += red[i];
    float inv = rsqrtf(tot * (1.0f / HD) + 1e-6f);
    const float* w = (h < 8) ? qw : kw;
    float y = x * inv * (1.0f + w[d]);
    sh[d] = y;
    __syncthreads();
    float rot = (d < 128) ? -sh[d + 128] : sh[d - 128];
    float r = y * cosb[(size_t)s * HD + d] + rot * sinb[(size_t)s * HD + d];
    if (h < 8) qo[((size_t)h * S_LEN + s) * HD + d] = f2bf(r);
    else       ko[((size_t)(h - 8) * S_LEN + s) * HD + d] = f2bf(r);
}

// ---------- flash attention with sliding window ----------
// grid: (S/64, NH), 128 threads = 4 waves; wave owns 16 query rows.
// Double-buffered K/V tiles; K staged with async global->LDS, V transposed
// through registers while WMMAs run on the current tile.
__global__ __launch_bounds__(128)
void attn_kernel(const unsigned short* __restrict__ Q,
                 const unsigned short* __restrict__ K,
                 const unsigned short* __restrict__ V,
                 unsigned short* __restrict__ O) {
    __shared__ alignas(16) unsigned short lK[2][32 * 256];   // [t][d]
    __shared__ alignas(16) unsigned short lVt[2][256 * 32];  // [d][t]
    __shared__ alignas(16) unsigned short lP[4][16 * 32];    // per-wave P staging

    const int tid = threadIdx.x;
    const int lane = tid & 31;
    const int wv = tid >> 5;
    const int hi = (lane >> 4) & 1;
    const int ln = lane & 15;
    const int h = blockIdx.y;
    const int kvh = h >> 1;                 // GQA rep=2
    const int q0 = blockIdx.x * 64;

    const unsigned short* Qh = Q + (size_t)h * S_LEN * HD;
    const unsigned short* Kh = K + (size_t)kvh * S_LEN * HD;
    const unsigned short* Vh = V + (size_t)kvh * S_LEN * HD;

    v16bf qf[8];
    const int qrow = q0 + wv * 16 + ln;
#pragma unroll
    for (int kb = 0; kb < 8; ++kb) {
        const unsigned short* p = Qh + (size_t)qrow * HD + kb * 32 + hi * 16;
        unpack8(*(const uint4*)p, qf[kb], 0);
        unpack8(*(const uint4*)(p + 8), qf[kb], 8);
    }

    const v8f vz = {0.f, 0.f, 0.f, 0.f, 0.f, 0.f, 0.f, 0.f};
    v8f o[16];
#pragma unroll
    for (int i = 0; i < 16; ++i) o[i] = vz;
    float row_m[8], row_l[8];
#pragma unroll
    for (int r = 0; r < 8; ++r) { row_m[r] = NEG_BIG; row_l[r] = 0.f; }

    int jmin = q0 - WINDOW;
    if (jmin < 0) jmin = 0;
    jmin &= ~31;
    const int jend = q0 + 63;
    const int nblk = (jend - jmin) / 32 + 1;

    // ---- prologue: stage first block into buffer 0 ----
    {
        const int kb0 = jmin;
#if HAS_ASYNC
#pragma unroll
        for (int ch = 0; ch < 8; ++ch) {
            int cid = ch * 128 + tid;
            int row = cid >> 5, cc = cid & 31;
            async_b128(Kh + (size_t)(kb0 + row) * HD + cc * 8,
                       &lK[0][row * 256 + cc * 8]);
        }
#else
#pragma unroll
        for (int ch = 0; ch < 8; ++ch) {
            int cid = ch * 128 + tid;
            int row = cid >> 5, cc = cid & 31;
            *(uint4*)&lK[0][row * 256 + cc * 8] =
                *(const uint4*)(Kh + (size_t)(kb0 + row) * HD + cc * 8);
        }
#endif
#pragma unroll
        for (int ch = 0; ch < 4; ++ch) {
            int cid = ch * 128 + tid;
            int rp = cid >> 5, cc = cid & 31;
            union { uint4 v; unsigned short u[8]; } t0, t1;
            t0.v = *(const uint4*)(Vh + (size_t)(kb0 + 2 * rp) * HD + cc * 8);
            t1.v = *(const uint4*)(Vh + (size_t)(kb0 + 2 * rp + 1) * HD + cc * 8);
#pragma unroll
            for (int j = 0; j < 8; ++j) {
                unsigned int w = (unsigned int)t0.u[j] | ((unsigned int)t1.u[j] << 16);
                *(unsigned int*)&lVt[0][(cc * 8 + j) * 32 + 2 * rp] = w;
            }
        }
#if HAS_ASYNC
        wait_async0();
#endif
        __syncthreads();
    }

    for (int bi = 0; bi < nblk; ++bi) {
        const int kb0 = jmin + bi * 32;
        const int cur = bi & 1, nxt = cur ^ 1;
        const bool have_next = (bi + 1 < nblk);
        uint4 rv0[4], rv1[4];
#if !HAS_ASYNC
        uint4 rk[8];
#endif
        if (have_next) {
            const int kn = kb0 + 32;
#if HAS_ASYNC
#pragma unroll
            for (int ch = 0; ch < 8; ++ch) {
                int cid = ch * 128 + tid;
                int row = cid >> 5, cc = cid & 31;
                async_b128(Kh + (size_t)(kn + row) * HD + cc * 8,
                           &lK[nxt][row * 256 + cc * 8]);
            }
#else
#pragma unroll
            for (int ch = 0; ch < 8; ++ch) {
                int cid = ch * 128 + tid;
                int row = cid >> 5, cc = cid & 31;
                rk[ch] = *(const uint4*)(Kh + (size_t)(kn + row) * HD + cc * 8);
            }
#endif
#pragma unroll
            for (int ch = 0; ch < 4; ++ch) {
                int cid = ch * 128 + tid;
                int rp = cid >> 5, cc = cid & 31;
                rv0[ch] = *(const uint4*)(Vh + (size_t)(kn + 2 * rp) * HD + cc * 8);
                rv1[ch] = *(const uint4*)(Vh + (size_t)(kn + 2 * rp + 1) * HD + cc * 8);
            }
        }

        // ---- scores on current tile: two 16-key tiles, 8 WMMA each over D ----
        v8f sc[2];
#pragma unroll
        for (int nt = 0; nt < 2; ++nt) {
            v8f acc = vz;
#pragma unroll
            for (int kb = 0; kb < 8; ++kb) {
                const unsigned short* p = &lK[cur][(nt * 16 + ln) * 256 + kb * 32 + hi * 16];
                v16bf b;
                unpack8(*(const uint4*)p, b, 0);
                unpack8(*(const uint4*)(p + 8), b, 8);
                acc = __builtin_amdgcn_wmma_f32_16x16x32_bf16(
                    false, qf[kb], false, b, (short)0, acc, false, false);
            }
            sc[nt] = acc;
        }
        // scale + sliding-window causal mask (C layout: row = r + hi*8, col = ln)
#pragma unroll
        for (int nt = 0; nt < 2; ++nt) {
            int j = kb0 + nt * 16 + ln;
#pragma unroll
            for (int r = 0; r < 8; ++r) {
                int i = q0 + wv * 16 + r + hi * 8;
                float sv = sc[nt][r] * SCALING;
                bool ok = (j <= i) && (i - j <= WINDOW);
                sc[nt][r] = ok ? sv : NEG_BIG;
            }
        }
        // online softmax: score row lives in 16 lanes of one half -> width-16 shfl
        float alpha[8];
#pragma unroll
        for (int r = 0; r < 8; ++r) {
            float mx = fmaxf(sc[0][r], sc[1][r]);
#pragma unroll
            for (int off = 8; off >= 1; off >>= 1) mx = fmaxf(mx, __shfl_xor(mx, off, 16));
            mx = fmaxf(mx, row_m[r]);
            alpha[r] = __expf(row_m[r] - mx);
            row_m[r] = mx;
            float p0 = __expf(sc[0][r] - mx);
            float p1 = __expf(sc[1][r] - mx);
            sc[0][r] = p0; sc[1][r] = p1;
            float sum = p0 + p1;
#pragma unroll
            for (int off = 8; off >= 1; off >>= 1) sum += __shfl_xor(sum, off, 16);
            row_l[r] = row_l[r] * alpha[r] + sum;
        }
        // restage P (C layout) -> A layout via per-wave LDS
        unsigned short* pw = &lP[wv][0];
#pragma unroll
        for (int nt = 0; nt < 2; ++nt)
#pragma unroll
            for (int r = 0; r < 8; ++r)
                pw[(r + hi * 8) * 32 + nt * 16 + ln] = f2bf(sc[nt][r]);
        asm volatile("s_wait_dscnt 0" ::: "memory");   // wave-local transpose fence
        v16bf pa;
        {
            const unsigned short* p = pw + ln * 32 + hi * 16;
            unpack8(*(const uint4*)p, pa, 0);
            unpack8(*(const uint4*)(p + 8), pa, 8);
        }
        // O = O*alpha + P @ V  (16 d-tiles)
#pragma unroll
        for (int dt = 0; dt < 16; ++dt) {
#pragma unroll
            for (int r = 0; r < 8; ++r) o[dt][r] *= alpha[r];
            const unsigned short* p = &lVt[cur][(dt * 16 + ln) * 32 + hi * 16];
            v16bf b;
            unpack8(*(const uint4*)p, b, 0);
            unpack8(*(const uint4*)(p + 8), b, 8);
            o[dt] = __builtin_amdgcn_wmma_f32_16x16x32_bf16(
                false, pa, false, b, (short)0, o[dt], false, false);
        }

        // ---- finish staging next tile ----
        if (have_next) {
#if !HAS_ASYNC
#pragma unroll
            for (int ch = 0; ch < 8; ++ch) {
                int cid = ch * 128 + tid;
                int row = cid >> 5, cc = cid & 31;
                *(uint4*)&lK[nxt][row * 256 + cc * 8] = rk[ch];
            }
#endif
#pragma unroll
            for (int ch = 0; ch < 4; ++ch) {
                int cid = ch * 128 + tid;
                int rp = cid >> 5, cc = cid & 31;
                union { uint4 v; unsigned short u[8]; } t0, t1;
                t0.v = rv0[ch]; t1.v = rv1[ch];
#pragma unroll
                for (int j = 0; j < 8; ++j) {
                    unsigned int w = (unsigned int)t0.u[j] | ((unsigned int)t1.u[j] << 16);
                    *(unsigned int*)&lVt[nxt][(cc * 8 + j) * 32 + 2 * rp] = w;
                }
            }
        }
#if HAS_ASYNC
        wait_async0();
#endif
        __syncthreads();
    }

    // epilogue: normalize and store bf16 attn[s][h*256 + d]
#pragma unroll
    for (int r = 0; r < 8; ++r) {
        float invl = 1.0f / row_l[r];
        int srow = q0 + wv * 16 + r + hi * 8;
        unsigned short* orow = O + (size_t)srow * ATTN_N + h * HD;
#pragma unroll
        for (int dt = 0; dt < 16; ++dt)
            orow[dt * 16 + ln] = f2bf(o[dt][r] * invl);
    }
}

// ---------- host launcher ----------
extern "C" void kernel_launch(void* const* d_in, const int* in_sizes, int n_in,
                              void* d_out, int out_size, void* d_ws, size_t ws_size,
                              hipStream_t stream) {
    (void)in_sizes; (void)n_in; (void)out_size; (void)ws_size;
    const float* hs   = (const float*)d_in[0];
    const float* cosb = (const float*)d_in[1];
    const float* sinb = (const float*)d_in[2];
    const float* wqkv = (const float*)d_in[3];
    const float* wo   = (const float*)d_in[4];
    const float* qw   = (const float*)d_in[5];
    const float* kw   = (const float*)d_in[6];
    float* out = (float*)d_out;

    const size_t HS_N   = (size_t)S_LEN * H_DIM;
    const size_t WQKV_N = (size_t)H_DIM * QKV_N;
    const size_t WO_N   = (size_t)ATTN_N * H_DIM;
    const size_t QKV_F  = (size_t)S_LEN * QKV_N;
    const size_t QH_N   = (size_t)NH  * S_LEN * HD;
    const size_t KV_N   = (size_t)NKV * S_LEN * HD;
    const size_t ATT_N  = (size_t)S_LEN * ATTN_N;

    char* ws = (char*)d_ws;
    size_t off = 0;
    auto take = [&](size_t bytes) -> char* {
        char* p = ws + off;
        off += bytes;
        off = (off + 255) & ~(size_t)255;
        return p;
    };
    unsigned short* hs_bf   = (unsigned short*)take(HS_N * 2);
    unsigned short* wqkv_bf = (unsigned short*)take(WQKV_N * 2);
    unsigned short* wo_bf   = (unsigned short*)take(WO_N * 2);
    float*          qkv_f   = (float*)take(QKV_F * 4);
    unsigned short* q_bf    = (unsigned short*)take(QH_N * 2);
    unsigned short* k_bf    = (unsigned short*)take(KV_N * 2);
    unsigned short* v_bf    = (unsigned short*)take(KV_N * 2);
    unsigned short* att_bf  = (unsigned short*)take(ATT_N * 2);

    cvt_f32_bf16<<<dim3((unsigned)((HS_N / 4 + 255) / 256)), 256, 0, stream>>>(hs, hs_bf, (long)(HS_N / 4));
    cvt_f32_bf16<<<dim3((unsigned)((WQKV_N / 4 + 255) / 256)), 256, 0, stream>>>(wqkv, wqkv_bf, (long)(WQKV_N / 4));
    cvt_f32_bf16<<<dim3((unsigned)((WO_N / 4 + 255) / 256)), 256, 0, stream>>>(wo, wo_bf, (long)(WO_N / 4));

    // qkv = hs @ w_qkv   [2048 x 2560] x [2560 x 4096]
    gemm_bf16<<<dim3(QKV_N / 128, S_LEN / 128), 256, 0, stream>>>(hs_bf, wqkv_bf, qkv_f, S_LEN, QKV_N, H_DIM);

    // RMS-norm + RoPE + bf16 repack of q/k/v
    normrope_kernel<<<dim3(S_LEN * (NH + 2 * NKV)), 256, 0, stream>>>(qkv_f, cosb, sinb, qw, kw, q_bf, k_bf, v_bf);

    // sliding-window flash attention
    attn_kernel<<<dim3(S_LEN / 64, NH), 128, 0, stream>>>(q_bf, k_bf, v_bf, att_bf);

    // out = attn @ w_o   [2048 x 2048] x [2048 x 2560]
    gemm_bf16<<<dim3(H_DIM / 128, S_LEN / 128), 256, 0, stream>>>(att_bf, wo_bf, out, S_LEN, H_DIM, ATTN_N);
}
